// LSMPool_16363825397821
// MI455X (gfx1250) — compile-verified, matrix operand order checked by reading, not run on hardware
//
#include <hip/hip_runtime.h>

// CDNA5 / gfx1250, wave32 only.
typedef __attribute__((ext_vector_type(16))) _Float16 v16h;
typedef __attribute__((ext_vector_type(8)))  _Float16 v8h;
typedef __attribute__((ext_vector_type(8)))  float    v8f;

#define T_STEPS 64
#define BATCH   256
#define IN_SZ   512
#define HID     2048
#define KTOT    (IN_SZ + HID)          // 2560, multiple of 32; IN_SZ boundary is 32-aligned
#define BH      ((size_t)BATCH * (size_t)HID)

// ---------------------------------------------------------------------------
// Elementwise f32 -> f16 conversion (W and x precision conversion, per call).
// ---------------------------------------------------------------------------
__global__ void lsm_cvt_f16(const float* __restrict__ src,
                            _Float16* __restrict__ dst, int n) {
  int i = blockIdx.x * blockDim.x + threadIdx.x;
  int stride = gridDim.x * blockDim.x;
  for (; i < n; i += stride) dst[i] = (_Float16)src[i];
}

// Zero the persistent membrane state and the t=0 hidden-spike buffer.
__global__ void lsm_zero_state(float* __restrict__ mem,
                               _Float16* __restrict__ spk, int n) {
  int i = blockIdx.x * blockDim.x + threadIdx.x;
  int stride = gridDim.x * blockDim.x;
  for (; i < n; i += stride) { mem[i] = 0.0f; spk[i] = (_Float16)0.0f; }
}

// ---------------------------------------------------------------------------
// One LIF reservoir timestep:
//   cur[b,n] = sum_k [x_t | spk_prev][b,k] * W[n,k]   (f16 WMMA, f32 accum)
//   mem' = mem + cur; spk = (mem' > 1); mem = mem' * (1 - spk)
// Block: 128 threads = 4 waves. Each wave -> 16(M) x 32(N) output strip
// (2x V_WMMA_F32_16X16X32_F16 tiles sharing one A fragment).
// Block tile: 32(M) x 64(N). Grid: (HID/64, BATCH/32) = (32, 8).
//
// Fragment layout assumption (ISA 7.12.2, wave32):
//   A 16x32 f16 : lane L -> row M=L&15; elems 0..7 = K=(L>>4)*8+[0..7],
//                 elems 8..15 = K=16+(L>>4)*8+[0..7]  (two b128 loads)
//   B 32x16 f16 : mirrored dual; lane L -> col N=L&15, same K
//                 pattern reading W row-major (two b128 loads)
//   C/D v8f     : elem r -> M = r + 8*(L>>4), N = L&15
// ---------------------------------------------------------------------------
__global__ __launch_bounds__(128) void lsm_step(
    const _Float16* __restrict__ Xt,      // (BATCH, IN_SZ)  f16, this timestep
    const _Float16* __restrict__ Wh,      // (HID, KTOT)     f16
    const _Float16* __restrict__ SpkIn,   // (BATCH, HID)    f16, prev hidden spikes
    _Float16* __restrict__ SpkOut,        // (BATCH, HID)    f16, next hidden spikes
    float* __restrict__ Mem,              // (BATCH, HID)    f32, persistent membrane
    float* __restrict__ SpkRec,           // (BATCH, HID)    f32 slice of d_out
    float* __restrict__ MemRec)           // (BATCH, HID)    f32 slice of d_out
{
  const int tid  = threadIdx.x;
  const int wave = tid >> 5;              // 0..3
  const int lane = tid & 31;
  const int half = lane >> 4;             // K-group select (0/1)
  const int mr   = lane & 15;             // row (A) / col (B) within 16-tile
  const int ko   = half * 8;

  const int mBase = blockIdx.y * 32 + (wave >> 1) * 16;   // batch row base
  const int nBase = blockIdx.x * 64 + (wave & 1) * 32;    // hidden col base

  union Frag { v16h f; v8h h[2]; };
  Frag a, b0, b1;
  v8f c0 = {};
  v8f c1 = {};

  // Per-lane row pointers (16B-aligned: all K offsets are multiples of 8 f16).
  const _Float16* aRowX = Xt    + (size_t)(mBase + mr) * IN_SZ;
  const _Float16* aRowS = SpkIn + (size_t)(mBase + mr) * HID;
  const _Float16* bRow0 = Wh + (size_t)(nBase + mr)      * KTOT;
  const _Float16* bRow1 = Wh + (size_t)(nBase + 16 + mr) * KTOT;

  // ---- K segment 1: input spikes x_t, K = 0..IN_SZ ----
#pragma unroll 2
  for (int k0 = 0; k0 < IN_SZ; k0 += 32) {
    a.h[0]  = *(const v8h*)(aRowX + k0 + ko);
    a.h[1]  = *(const v8h*)(aRowX + k0 + 16 + ko);
    b0.h[0] = *(const v8h*)(bRow0 + k0 + ko);
    b0.h[1] = *(const v8h*)(bRow0 + k0 + 16 + ko);
    b1.h[0] = *(const v8h*)(bRow1 + k0 + ko);
    b1.h[1] = *(const v8h*)(bRow1 + k0 + 16 + ko);
    __builtin_prefetch(bRow0 + k0 + 64, 0, 0);   // global_prefetch_b8, L2 stream
    c0 = __builtin_amdgcn_wmma_f32_16x16x32_f16(false, a.f, false, b0.f,
                                                (short)0, c0, false, false);
    c1 = __builtin_amdgcn_wmma_f32_16x16x32_f16(false, a.f, false, b1.f,
                                                (short)0, c1, false, false);
  }

  // ---- K segment 2: recurrent hidden spikes, K = IN_SZ..KTOT ----
#pragma unroll 2
  for (int k0 = 0; k0 < HID; k0 += 32) {
    a.h[0]  = *(const v8h*)(aRowS + k0 + ko);
    a.h[1]  = *(const v8h*)(aRowS + k0 + 16 + ko);
    b0.h[0] = *(const v8h*)(bRow0 + IN_SZ + k0 + ko);
    b0.h[1] = *(const v8h*)(bRow0 + IN_SZ + k0 + 16 + ko);
    b1.h[0] = *(const v8h*)(bRow1 + IN_SZ + k0 + ko);
    b1.h[1] = *(const v8h*)(bRow1 + IN_SZ + k0 + 16 + ko);
    __builtin_prefetch(bRow0 + IN_SZ + k0 + 64, 0, 0);
    c0 = __builtin_amdgcn_wmma_f32_16x16x32_f16(false, a.f, false, b0.f,
                                                (short)0, c0, false, false);
    c1 = __builtin_amdgcn_wmma_f32_16x16x32_f16(false, a.f, false, b1.f,
                                                (short)0, c1, false, false);
  }

  // ---- LIF epilogue: beta = 1.0, threshold = 1.0, reset-to-zero ----
  const int mOff = mBase + half * 8;
#pragma unroll
  for (int r = 0; r < 8; ++r) {
    const size_t row = (size_t)(mOff + r) * HID;
    {
      const size_t idx = row + (size_t)(nBase + mr);
      float mv = Mem[idx] + c0[r];
      float s  = (mv > 1.0f) ? 1.0f : 0.0f;
      float mn = mv * (1.0f - s);
      SpkRec[idx] = s;
      MemRec[idx] = mn;
      Mem[idx]    = mn;
      SpkOut[idx] = (_Float16)s;
    }
    {
      const size_t idx = row + (size_t)(nBase + 16 + mr);
      float mv = Mem[idx] + c1[r];
      float s  = (mv > 1.0f) ? 1.0f : 0.0f;
      float mn = mv * (1.0f - s);
      SpkRec[idx] = s;
      MemRec[idx] = mn;
      Mem[idx]    = mn;
      SpkOut[idx] = (_Float16)s;
    }
  }
}

// ---------------------------------------------------------------------------
// Host-side orchestration: precision-convert once, then 64 dependent steps.
// Workspace layout (all sizes 256B-aligned by construction):
//   Wh   : HID*KTOT f16            = 10.49 MB
//   Xh   : T*BATCH*IN_SZ f16       = 16.78 MB
//   Spk  : 2 * BATCH*HID f16       =  2.10 MB  (double buffer)
//   Mem  : BATCH*HID f32           =  2.10 MB
// ---------------------------------------------------------------------------
extern "C" void kernel_launch(void* const* d_in, const int* in_sizes, int n_in,
                              void* d_out, int out_size, void* d_ws, size_t ws_size,
                              hipStream_t stream) {
  (void)in_sizes; (void)n_in; (void)out_size; (void)ws_size;

  const float* x = (const float*)d_in[0];   // (T, B, IN) f32 0/1 spikes
  const float* W = (const float*)d_in[1];   // (H, IN+H)  f32
  float* out = (float*)d_out;               // spk_rec ‖ mem_rec, each T*B*H f32

  char* ws = (char*)d_ws;
  size_t off = 0;
  _Float16* Wh  = (_Float16*)(ws + off); off += (size_t)HID * KTOT * sizeof(_Float16);
  _Float16* Xh  = (_Float16*)(ws + off); off += (size_t)T_STEPS * BATCH * IN_SZ * sizeof(_Float16);
  _Float16* Spk = (_Float16*)(ws + off); off += 2 * BH * sizeof(_Float16);
  float*    Mem = (float*)   (ws + off); off += BH * sizeof(float);

  // Precision conversion + state init (deterministic, every call).
  const int nW = HID * KTOT;
  const int nX = T_STEPS * BATCH * IN_SZ;
  lsm_cvt_f16<<<1024, 256, 0, stream>>>(W, Wh, nW);
  lsm_cvt_f16<<<1024, 256, 0, stream>>>(x, Xh, nX);
  lsm_zero_state<<<1024, 256, 0, stream>>>(Mem, Spk, (int)BH);

  // Sequential recurrence: one launch per timestep (stream-ordered).
  dim3 grid(HID / 64, BATCH / 32);
  for (int t = 0; t < T_STEPS; ++t) {
    lsm_step<<<grid, 128, 0, stream>>>(
        Xh + (size_t)t * BATCH * IN_SZ,
        Wh,
        Spk + (size_t)(t & 1) * BH,
        Spk + (size_t)((t + 1) & 1) * BH,
        Mem,
        out + (size_t)t * BH,
        out + (size_t)T_STEPS * BH + (size_t)t * BH);
  }
}
